// AttentionBlock_85246510891450
// MI455X (gfx1250) — matrix operand order repaired and checked by reading, not compiled, not run on hardware
//
#include <hip/hip_runtime.h>
#include <hip/hip_bf16.h>
#include <math.h>

typedef _Float16 half_t;
typedef __attribute__((ext_vector_type(16))) _Float16 v16h;
typedef __attribute__((ext_vector_type(8)))  _Float16 v8h;
typedef __attribute__((ext_vector_type(8)))  float    v8f;

#define BATCH 16
#define CCH   512
#define NPOS  1024
#define NGRP  32

// ---------------------------------------------------------------------------
// WMMA-B fragment-order packing for a KxN f16 matrix, tiles of 32(K) x 16(N).
// Tile (k/32, n/16) occupies 512 contiguous halves; within a tile lane
// l = ((k>>4)&1)*16 + (n&15) owns halves [l*16, l*16+16) = K rows (k&15).
// A wave's B fragment is then ONE contiguous 32-byte load per lane, and the
// fragment for the adjacent 16 columns sits exactly 512 halves later.
// ---------------------------------------------------------------------------
__device__ __forceinline__ size_t bpack_off(int k, int n, int nt16) {
    return ((size_t)(k >> 5) * nt16 + (n >> 4)) * 512
         + (size_t)((((k >> 4) & 1) * 16 + (n & 15)) * 16 + (k & 15));
}

// ---------------------------------------------------------------------------
// Convert the four 512x512 f32 weights to f16 (A operands stay row-major).
// ---------------------------------------------------------------------------
__global__ void convert_w_kernel(const float* __restrict__ wq, const float* __restrict__ wk,
                                 const float* __restrict__ wv, const float* __restrict__ wp,
                                 half_t* __restrict__ out) {
    int i = blockIdx.x * 256 + threadIdx.x;            // 262144 elements each
    out[i]            = (half_t)wq[i];
    out[262144 + i]   = (half_t)wk[i];
    out[524288 + i]   = (half_t)wv[i];
    out[786432 + i]   = (half_t)wp[i];
}

// ---------------------------------------------------------------------------
// GroupNorm: one block per (batch, group); output hn in B-packed f16 layout.
// ---------------------------------------------------------------------------
__global__ void groupnorm_kernel(const float* __restrict__ x, const float* __restrict__ gamma,
                                 const float* __restrict__ beta, half_t* __restrict__ hn) {
    __shared__ float red[256], red2[256];
    int b = blockIdx.x >> 5, g = blockIdx.x & 31;
    const float* xg = x + ((size_t)b * CCH + (size_t)g * 16) * NPOS;   // 16384 floats
    float s = 0.f, s2 = 0.f;
    for (int t = threadIdx.x; t < 16 * NPOS; t += 256) {
        float v = xg[t]; s += v; s2 += v * v;
    }
    red[threadIdx.x] = s; red2[threadIdx.x] = s2; __syncthreads();
    for (int off = 128; off > 0; off >>= 1) {
        if (threadIdx.x < off) {
            red[threadIdx.x]  += red[threadIdx.x + off];
            red2[threadIdx.x] += red2[threadIdx.x + off];
        }
        __syncthreads();
    }
    const float cnt_inv = 1.f / (16.f * NPOS);
    float mean = red[0] * cnt_inv;
    float var  = red2[0] * cnt_inv - mean * mean;
    float inv  = rsqrtf(var + 1e-5f);
    half_t* hb = hn + (size_t)b * CCH * NPOS;
    for (int t = threadIdx.x; t < 16 * NPOS; t += 256) {
        int cl = t >> 10, n = t & 1023;
        int ch = g * 16 + cl;
        float v = ((xg[t] - mean) * inv) * gamma[ch] + beta[ch];
        hb[bpack_off(ch, n, NPOS / 16)] = (half_t)v;
    }
}

// ---------------------------------------------------------------------------
// Batched WMMA GEMM:  C[b] = alpha * A[b](MxK row-major) * B[b](KxN B-PACKED)
//   OUTMODE 0: f16 out, element (m,n) at m*crs + n*ccs   (row-major / transposed)
//   OUTMODE 1: f16 out in B-packed layout (cnt = N/16 of the matrix as consumed)
//   OUTMODE 2: f32 out [m][n] + bias + residual (final projection)
// Wave tile 64x32: 4 A fragments reused over 2 B fragments -> 8 WMMA per
// 12 b128 loads per k-step.  Block tile 128x128 (8 waves, 2x4).
// M%128==0, N%128==0, K%32==0: no bounds checks, EXEC all-ones around WMMA.
// ---------------------------------------------------------------------------
template <int OUTMODE, int BIAS>
__global__ __launch_bounds__(256)
void wmma_gemm_kernel(const half_t* __restrict__ A, long sAb, int lda,
                      const half_t* __restrict__ B, long sBb, int bnt,
                      void* __restrict__ C, long sCb, int crs, int ccs, int cnt,
                      const float* __restrict__ bias,
                      const float* __restrict__ residual, long sRb,
                      float alpha, int K) {
    const int bz   = blockIdx.z;
    const half_t* Ab = A + (long)bz * sAb;
    const half_t* Bb = B + (long)bz * sBb;
    const int wave = threadIdx.x >> 5;
    const int lane = threadIdx.x & 31;
    const int m0   = blockIdx.x * 128 + (wave >> 2) * 64;
    const int n0   = blockIdx.y * 128 + (wave & 3) * 32;
    const int r    = lane & 15;
    const int hh   = lane >> 4;

    // B: two adjacent 32-byte contiguous loads per lane per k-step
    const half_t* bptr  = Bb + (size_t)(n0 >> 4) * 512 + (size_t)lane * 16;
    const long    bstep = (long)bnt * 512;        // advance one 32-row k-tile

    v8f acc0[4] = {};   // columns n0 .. n0+15
    v8f acc1[4] = {};   // columns n0+16 .. n0+31
    for (int k = 0; k < K; k += 32) {
        v16h bf0 = *(const v16h*)bptr;
        v16h bf1 = *(const v16h*)(bptr + 512);
        bptr += bstep;
        const half_t* ap = Ab + (long)(m0 + r) * lda + k;
        v16h af[4];
#pragma unroll
        for (int t = 0; t < 4; ++t) {
            const half_t* apt = ap + (long)t * 16 * lda;
            v8h alo = *(const v8h*)(apt + hh * 16);       // K = hh*16 + 0..7
            v8h ahi = *(const v8h*)(apt + hh * 16 + 8);   // K = hh*16 + 8..15
            af[t] = __builtin_shufflevector(alo, ahi, 0, 1, 2, 3, 4, 5, 6, 7,
                                            8, 9, 10, 11, 12, 13, 14, 15);
        }
#pragma unroll
        for (int t = 0; t < 4; ++t)
            acc0[t] = __builtin_amdgcn_wmma_f32_16x16x32_f16(
                false, af[t], false, bf0, (short)0, acc0[t], false, false);
#pragma unroll
        for (int t = 0; t < 4; ++t)
            acc1[t] = __builtin_amdgcn_wmma_f32_16x16x32_f16(
                false, af[t], false, bf1, (short)0, acc1[t], false, false);
    }

    // ---- epilogue: VGPR v -> M = t*16 + hh*8 + v, N = n0 + j*16 + r
#pragma unroll
    for (int t = 0; t < 4; ++t) {
        v8f bv = {};
        if (BIAS) bv = *(const v8f*)(bias + m0 + t * 16 + hh * 8);  // one aligned 32B load
#pragma unroll
        for (int j = 0; j < 2; ++j) {
            const v8f& a = j ? acc1[t] : acc0[t];
            const int n = n0 + j * 16 + r;
#pragma unroll
            for (int v = 0; v < 8; ++v) {
                int m = m0 + t * 16 + hh * 8 + v;
                float val = a[v] * alpha;
                if (BIAS) val += bv[v];
                if (OUTMODE == 0) {
                    ((half_t*)C + (long)bz * sCb)[(long)m * crs + (long)n * ccs] = (half_t)val;
                } else if (OUTMODE == 1) {
                    ((half_t*)C + (long)bz * sCb)[bpack_off(m, n, cnt)] = (half_t)val;
                } else {
                    float* Co = (float*)C + (long)bz * sCb;
                    val += residual[(long)bz * sRb + (long)m * NPOS + n];
                    Co[(long)m * NPOS + n] = val;
                }
            }
        }
    }
}

// ---------------------------------------------------------------------------
// Row softmax of S[b][i][:] (f16 logits, f32 math), P written TRANSPOSED and
// B-PACKED (element (j,i) as a 1024x1024 K x N matrix for the AV GEMM).
// ---------------------------------------------------------------------------
__global__ void softmax_kernel(const half_t* __restrict__ S, half_t* __restrict__ Pt) {
    __shared__ float red[256];
    int b = blockIdx.x >> 10, i = blockIdx.x & 1023;
    const half_t* row = S + ((size_t)b * NPOS + i) * NPOS;
    float vals[4];
    float lmax = -1e30f;
#pragma unroll
    for (int u = 0; u < 4; ++u) {
        float v = (float)row[threadIdx.x + u * 256];
        vals[u] = v; lmax = fmaxf(lmax, v);
    }
    red[threadIdx.x] = lmax; __syncthreads();
    for (int off = 128; off > 0; off >>= 1) {
        if (threadIdx.x < off) red[threadIdx.x] = fmaxf(red[threadIdx.x], red[threadIdx.x + off]);
        __syncthreads();
    }
    float m = red[0]; __syncthreads();
    float ls = 0.f;
#pragma unroll
    for (int u = 0; u < 4; ++u) { vals[u] = __expf(vals[u] - m); ls += vals[u]; }
    red[threadIdx.x] = ls; __syncthreads();
    for (int off = 128; off > 0; off >>= 1) {
        if (threadIdx.x < off) red[threadIdx.x] += red[threadIdx.x + off];
        __syncthreads();
    }
    float inv = 1.f / red[0];
    half_t* Ptb = Pt + (size_t)b * NPOS * NPOS;
#pragma unroll
    for (int u = 0; u < 4; ++u) {
        int j = threadIdx.x + u * 256;
        Ptb[bpack_off(j, i, NPOS / 16)] = (half_t)(vals[u] * inv);
    }
}

// ---------------------------------------------------------------------------
extern "C" void kernel_launch(void* const* d_in, const int* in_sizes, int n_in,
                              void* d_out, int out_size, void* d_ws, size_t ws_size,
                              hipStream_t stream) {
    const float* x     = (const float*)d_in[0];
    const float* gamma = (const float*)d_in[1];
    const float* beta  = (const float*)d_in[2];
    const float* wq    = (const float*)d_in[3];
    const float* bq    = (const float*)d_in[4];
    const float* wk    = (const float*)d_in[5];
    const float* bk    = (const float*)d_in[6];
    const float* wv    = (const float*)d_in[7];
    const float* bv    = (const float*)d_in[8];
    const float* wp    = (const float*)d_in[9];
    const float* bp    = (const float*)d_in[10];

    // f16 workspace layout (element offsets)
    half_t* ws   = (half_t*)d_ws;
    half_t* wq16 = ws;                       // 262144 each, row-major (A operands)
    half_t* wk16 = ws + 262144;
    half_t* wv16 = ws + 524288;
    half_t* wp16 = ws + 786432;
    half_t* hn   = ws + 1048576;             // [b] 512x1024  B-PACKED
    half_t* qT   = hn + 8388608;             // [b][n][c] row-major (A of scores)
    half_t* kb   = qT + 8388608;             // [b] 512x1024  B-PACKED
    half_t* vb   = kb + 8388608;             // [b][c][n] row-major (A of AV)
    half_t* S    = vb + 8388608;             // [b][i][j] row-major f16
    half_t* Pt   = S  + 16777216;            // [b] 1024x1024 (j,i) B-PACKED
    half_t* O    = Pt + 16777216;            // [b] 512x1024  B-PACKED

    const long CN = (long)CCH * NPOS;        // 524288
    const long NN = (long)NPOS * NPOS;       // 1048576
    const int  NT = NPOS / 16;               // 64
    const float scale = 0.044194173824159216f;  // 512^-0.5

    convert_w_kernel<<<1024, 256, 0, stream>>>(wq, wk, wv, wp, ws);
    groupnorm_kernel<<<BATCH * NGRP, 256, 0, stream>>>(x, gamma, beta, hn);

    dim3 gProj(CCH / 128, NPOS / 128, BATCH);   // M=512, N=1024
    dim3 gScor(NPOS / 128, NPOS / 128, BATCH);  // M=1024, N=1024

    // Q = Wq*H + bq -> qT row-major [n][c]  (crs=1, ccs=512)
    wmma_gemm_kernel<0, 1><<<gProj, 256, 0, stream>>>(wq16, 0, CCH, hn, CN, NT,
                                                      qT, CN, 1, CCH, 0, bq, nullptr, 0, 1.f, CCH);
    // K = Wk*H + bk -> B-packed
    wmma_gemm_kernel<1, 1><<<gProj, 256, 0, stream>>>(wk16, 0, CCH, hn, CN, NT,
                                                      kb, CN, 0, 0, NT, bk, nullptr, 0, 1.f, CCH);
    // V = Wv*H + bv -> row-major [c][n]
    wmma_gemm_kernel<0, 1><<<gProj, 256, 0, stream>>>(wv16, 0, CCH, hn, CN, NT,
                                                      vb, CN, NPOS, 1, 0, bv, nullptr, 0, 1.f, CCH);
    // S = scale * Q^T K  -> row-major [i][j]
    wmma_gemm_kernel<0, 0><<<gScor, 256, 0, stream>>>(qT, CN, CCH, kb, CN, NT,
                                                      S, NN, NPOS, 1, 0, nullptr, nullptr, 0, scale, CCH);
    // softmax rows -> Pt (j,i) B-packed
    softmax_kernel<<<BATCH * NPOS, 256, 0, stream>>>(S, Pt);
    // O = V * P^T  -> B-packed,  K = 1024
    wmma_gemm_kernel<1, 0><<<gProj, 256, 0, stream>>>(vb, CN, NPOS, Pt, NN, NT,
                                                      O, CN, 0, 0, NT, nullptr, nullptr, 0, 1.f, NPOS);
    // out = Wp*O + bp + x   (f32 out + residual)
    wmma_gemm_kernel<2, 1><<<gProj, 256, 0, stream>>>(wp16, 0, CCH, O, CN, NT,
                                                      d_out, CN, 0, 0, 0, bp, x, CN, 1.f, CCH);
}
